// JointEBM_17334488007244
// MI455X (gfx1250) — matrix operand order repaired and checked by reading, not compiled, not run on hardware
//
#include <hip/hip_runtime.h>

// ---------------------------------------------------------------------------
// Joint-EBM inner-loop gradient descent, CDNA5 (gfx1250), wave32 WMMA bf16.
// B=65536, DX=256, DY=64, H=512, K=4, steps=20, LR=0.1
// ---------------------------------------------------------------------------

typedef __attribute__((ext_vector_type(16))) __bf16 v16bf;
typedef __attribute__((ext_vector_type(8)))  __bf16 v8bf;
typedef __attribute__((ext_vector_type(8)))  float  v8f;

#define DXc 256
#define DYc 64
#define Hc  512
#define Bc  65536
#define MT  128      // batch rows per block (8 waves x 16-row WMMA tiles)
#define LRc 0.1f

union AFrag { v16bf v; v8bf h[2]; };

static __device__ inline unsigned short f2bf(float f) {
  unsigned u = __builtin_bit_cast(unsigned, f);
  return (unsigned short)((u + 0x7FFFu + ((u >> 16) & 1u)) >> 16);  // RNE
}
static __device__ inline float bf2f(unsigned short h) {
  return __builtin_bit_cast(float, (unsigned)h << 16);
}

// A fragment (16x32 bf16, M x K) from a row-major bf16 tile in LDS.
// Lane l<16 : row M=l,    K = [0..7] and [16..23]
// Lane l>=16: row M=l-16, K = [8..15] and [24..31]
static __device__ inline v16bf load_a(const unsigned short* base, int stride,
                                      int k0, int lane) {
  const unsigned short* p = base + (lane & 15) * stride + k0 + ((lane >> 4) * 8);
  AFrag f;
  f.h[0] = *(const v8bf*)(p);
  f.h[1] = *(const v8bf*)(p + 16);
  return f.v;
}

// B fragment (32x16 bf16, K x N) from weights pre-stored as [n][k] row-major.
// Lane l<16 : col N=n0+l,    K = [0..7] and [16..23]
// Lane l>=16: col N=n0+l-16, K = [8..15] and [24..31]
static __device__ inline v16bf load_b(const unsigned short* w,
                                      int kdim, int n0, int k0, int lane) {
  const unsigned short* p = w + (size_t)(n0 + (lane & 15)) * kdim + k0
                              + ((lane >> 4) * 8);
  AFrag f;
  f.h[0] = *(const v8bf*)(p);
  f.h[1] = *(const v8bf*)(p + 16);
  return f.v;
}

static __device__ inline v8f wmma_bf16(v16bf a, v16bf b, v8f c) {
  return __builtin_amdgcn_wmma_f32_16x16x32_bf16(false, a, false, b,
                                                 (short)0, c, false, false);
}

// ---------------------------------------------------------------------------
// Weight preprocessing: fp32 -> bf16, optional transpose into [n][k] layout.
// ---------------------------------------------------------------------------
__global__ void cvt_t(const float* __restrict__ src, unsigned short* __restrict__ dst,
                      int R, int C) {
  int i = blockIdx.x * 256 + threadIdx.x;          // src is [R][C] row-major
  if (i < R * C) {
    int r = i / C, c = i % C;
    dst[c * R + r] = f2bf(src[i]);                 // dst is [C][R]
  }
}
__global__ void cvt(const float* __restrict__ src, unsigned short* __restrict__ dst,
                    int N) {
  int i = blockIdx.x * 256 + threadIdx.x;
  if (i < N) dst[i] = f2bf(src[i]);
}

// ---------------------------------------------------------------------------
// Main kernel: 20 descent steps fully fused, activations resident in 320KB LDS.
// ---------------------------------------------------------------------------
extern "C" __global__ __launch_bounds__(256, 1)
void ebm_descent(const float* __restrict__ x, const int* __restrict__ t,
                 const float* __restrict__ b1, const float* __restrict__ b2,
                 const unsigned short* __restrict__ W1t,  // [512][320] = W1^T
                 const unsigned short* __restrict__ W2t,  // [512][512] = W2^T
                 const unsigned short* __restrict__ W2r,  // [512][512] = W2 row-major
                 const unsigned short* __restrict__ W1y,  // [64][512]  = W1 rows 256..319
                 const unsigned short* __restrict__ W3t,  // [4][512]   = W3^T
                 const int* __restrict__ steps_p,
                 float* __restrict__ out) {
  extern __shared__ unsigned short smem[];
  unsigned short* xs   = smem;                       // [128][256] bf16, 64KB
  unsigned short* bufA = smem + MT * DXc;            // [128][512] h1 -> g1, 128KB
  unsigned short* bufB = smem + MT * DXc + MT * Hc;  // [128][512] g2, 128KB
  unsigned short* ybf  = bufB;                       // [128][64] y bf16 (aliases g2 rows 0..15)

  const int tid   = threadIdx.x;
  const int lane  = tid & 31;
  const int wv    = tid >> 5;
  const int mw    = wv * 16;                 // wave's local row base
  const int mblk  = blockIdx.x * MT;
  const int half  = lane >> 4;
  const int nlane = lane & 15;

  // Stage x tile into LDS as bf16 (reused across all steps).
  for (int i = tid; i < MT * DXc; i += 256) {
    int r = i >> 8, c = i & 255;
    xs[i] = f2bf(x[(size_t)(mblk + r) * DXc + c]);
  }
  // y0 = 0 (bf16 shadow).
  for (int i = tid; i < MT * DYc; i += 256) ybf[i] = 0;

  // Pack this lane's 8 class labels (t in [0,4)) into 2-bit fields.
  unsigned tp = 0;
  for (int r = 0; r < 8; ++r) {
    int m = mblk + mw + r + 8 * half;        // C-layout row of accumulator reg r
    tp |= ((unsigned)t[m] & 3u) << (2 * r);
  }

  v8f yacc[4];                               // y master copy, f32, C layout
  for (int nt = 0; nt < 4; ++nt)
    for (int r = 0; r < 8; ++r) yacc[nt][r] = 0.f;

  __syncthreads();

  // Opaque zero: makes weight addresses loop-variant (blocks the cross-step
  // LICM that spilled 300+ B fragments to scratch in round 1) while keeping
  // pointer provenance on the __restrict__ kernargs, so address-space
  // inference still emits global_load_b128 (LOADcnt only) instead of the
  // flat_load_b128 (LOADcnt+DScnt, LDS-path contention) seen in round 2.
  int sz = 0;

  const int ns = steps_p[0];
  for (int s = 0; s < ns; ++s) {
    asm volatile("" : "+s"(sz));
    const unsigned short* w1t = W1t + sz;
    const unsigned short* w2t = W2t + sz;
    const unsigned short* w2r = W2r + sz;
    const unsigned short* w1y = W1y + sz;
    const unsigned short* w3t = W3t + sz;
    const float* pb1 = b1 + sz;
    const float* pb2 = b2 + sz;

    // ---- stage 1: h1 = relu([x,y] @ W1 + b1) -> bufA -------------------
    for (int nt = 0; nt < Hc / 16; ++nt) {
      float bias = pb1[nt * 16 + nlane];
      v8f acc;
      for (int r = 0; r < 8; ++r) acc[r] = bias;
      v16bf a = load_a(xs + mw * DXc, DXc, 0, lane);
      v16bf b = load_b(w1t, DXc + DYc, nt * 16, 0, lane);
#pragma unroll
      for (int kt = 0; kt < 10; ++kt) {      // K = 320 = 256(x) + 64(y)
        v16bf ca = a, cb = b;
        if (kt + 1 < 10) {                   // prefetch next fragments
          a = (kt + 1 < 8) ? load_a(xs + mw * DXc, DXc, (kt + 1) * 32, lane)
                           : load_a(ybf + mw * DYc, DYc, (kt + 1 - 8) * 32, lane);
          b = load_b(w1t, DXc + DYc, nt * 16, (kt + 1) * 32, lane);
        }
        acc = wmma_bf16(ca, cb, acc);
      }
      for (int r = 0; r < 8; ++r) {
        float v = fmaxf(acc[r], 0.f);
        bufA[(mw + r + 8 * half) * Hc + nt * 16 + nlane] = f2bf(v);
      }
    }
    __syncthreads();  // everyone done reading ybf; bufB may be clobbered

    // ---- stage 2: a2 = h1 @ W2 + b2 ; g2 = (a2>0)?W3[:,t]:0 -> bufB ----
    for (int nt = 0; nt < Hc / 16; ++nt) {
      int n = nt * 16 + nlane;
      float bias = pb2[n];
      v8f acc;
      for (int r = 0; r < 8; ++r) acc[r] = bias;
      v16bf a = load_a(bufA + mw * Hc, Hc, 0, lane);
      v16bf b = load_b(w2t, Hc, nt * 16, 0, lane);
#pragma unroll
      for (int kt = 0; kt < Hc / 32; ++kt) {
        v16bf ca = a, cb = b;
        if (kt + 1 < Hc / 32) {
          a = load_a(bufA + mw * Hc, Hc, (kt + 1) * 32, lane);
          b = load_b(w2t, Hc, nt * 16, (kt + 1) * 32, lane);
        }
        acc = wmma_bf16(ca, cb, acc);
      }
      unsigned short w3v[4];
      for (int j = 0; j < 4; ++j) w3v[j] = w3t[j * Hc + n];
      for (int r = 0; r < 8; ++r) {
        unsigned sel = (tp >> (2 * r)) & 3u;
        bufB[(mw + r + 8 * half) * Hc + n] =
            (acc[r] > 0.f) ? w3v[sel] : (unsigned short)0;
      }
    }
    // bufA/bufB rows are wave-private from here -> no barrier needed

    // ---- stage 3: g1 = (h1>0) ? (g2 @ W2^T) : 0, in-place into bufA ----
    for (int nt = 0; nt < Hc / 16; ++nt) {
      v8f acc;
      for (int r = 0; r < 8; ++r) acc[r] = 0.f;
      v16bf a = load_a(bufB + mw * Hc, Hc, 0, lane);
      v16bf b = load_b(w2r, Hc, nt * 16, 0, lane);
#pragma unroll
      for (int kt = 0; kt < Hc / 32; ++kt) {
        v16bf ca = a, cb = b;
        if (kt + 1 < Hc / 32) {
          a = load_a(bufB + mw * Hc, Hc, (kt + 1) * 32, lane);
          b = load_b(w2r, Hc, nt * 16, (kt + 1) * 32, lane);
        }
        acc = wmma_bf16(ca, cb, acc);
      }
      for (int r = 0; r < 8; ++r) {
        int idx = (mw + r + 8 * half) * Hc + nt * 16 + nlane;
        unsigned short h = bufA[idx];
        bufA[idx] = (bf2f(h) > 0.f) ? f2bf(acc[r]) : (unsigned short)0;
      }
    }
    __syncthreads();  // all waves done reading g2 before ybf overwrites it

    // ---- stage 4: dy = g1 @ W1_y^T ; y -= LR*dy ; refresh bf16 shadow --
    for (int nt = 0; nt < DYc / 16; ++nt) {
      v8f acc;
      for (int r = 0; r < 8; ++r) acc[r] = 0.f;
      v16bf a = load_a(bufA + mw * Hc, Hc, 0, lane);
      v16bf b = load_b(w1y, Hc, nt * 16, 0, lane);
#pragma unroll
      for (int kt = 0; kt < Hc / 32; ++kt) {
        v16bf ca = a, cb = b;
        if (kt + 1 < Hc / 32) {
          a = load_a(bufA + mw * Hc, Hc, (kt + 1) * 32, lane);
          b = load_b(w1y, Hc, nt * 16, (kt + 1) * 32, lane);
        }
        acc = wmma_bf16(ca, cb, acc);
      }
      for (int r = 0; r < 8; ++r) {
        yacc[nt][r] -= LRc * acc[r];
        ybf[(mw + r + 8 * half) * DYc + nt * 16 + nlane] = f2bf(yacc[nt][r]);
      }
    }
    // next step's stage-1 ybf reads are wave-private; the barrier after
    // stage 1 covers the cross-wave ybf-vs-g2 alias.
  }

  // Emit y_final (f32).
  for (int nt = 0; nt < 4; ++nt)
    for (int r = 0; r < 8; ++r)
      out[(size_t)(mblk + mw + r + 8 * half) * DYc + nt * 16 + nlane] = yacc[nt][r];
}

// ---------------------------------------------------------------------------
// Host-side launch (graph-capture safe: only kernel launches on `stream`).
// ---------------------------------------------------------------------------
extern "C" void kernel_launch(void* const* d_in, const int* in_sizes, int n_in,
                              void* d_out, int out_size, void* d_ws, size_t ws_size,
                              hipStream_t stream) {
  const float* x  = (const float*)d_in[0];
  const int*   t  = (const int*)d_in[1];
  const float* W1 = (const float*)d_in[2];   // [320][512]
  const float* b1 = (const float*)d_in[3];
  const float* W2 = (const float*)d_in[4];   // [512][512]
  const float* b2 = (const float*)d_in[5];
  const float* W3 = (const float*)d_in[6];   // [512][4]
  // d_in[7] = b3: cancels out of grad_y, never needed.
  const int* steps = (const int*)d_in[8];

  // bf16 weight workspace (~1.45 MB total, stays hot in L2).
  unsigned short* ws  = (unsigned short*)d_ws;
  unsigned short* W1t = ws;                          // [512][320]
  unsigned short* W2t = W1t + 512 * 320;             // [512][512]
  unsigned short* W2r = W2t + 512 * 512;             // [512][512]
  unsigned short* W1y = W2r + 512 * 512;             // [64][512]
  unsigned short* W3t = W1y + 64 * 512;              // [4][512]

  cvt_t<<<(320 * 512 + 255) / 256, 256, 0, stream>>>(W1, W1t, 320, 512);
  cvt_t<<<(512 * 512 + 255) / 256, 256, 0, stream>>>(W2, W2t, 512, 512);
  cvt  <<<(512 * 512 + 255) / 256, 256, 0, stream>>>(W2, W2r, 512 * 512);
  cvt  <<<(64 * 512 + 255) / 256, 256, 0, stream>>>(W1 + 256 * 512, W1y, 64 * 512);
  cvt_t<<<(512 * 4 + 255) / 256, 256, 0, stream>>>(W3, W3t, 512, 4);

  const size_t shmem = (size_t)(MT * DXc + 2 * MT * Hc) * sizeof(unsigned short); // 320 KB
  (void)hipFuncSetAttribute(reinterpret_cast<const void*>(ebm_descent),
                            hipFuncAttributeMaxDynamicSharedMemorySize, (int)shmem);

  ebm_descent<<<Bc / MT, 256, shmem, stream>>>(x, t, b1, b2, W1t, W2t, W2r, W1y,
                                               W3t, steps, (float*)d_out);
}
